// SwinMLPBlock_2619930051360
// MI455X (gfx1250) — matrix operand order repaired and checked by reading, not compiled, not run on hardware
//
#include <hip/hip_runtime.h>

// ---------------------------------------------------------------------------
// SwinMLP block for MI455X (gfx1250, wave32, WMMA, async-to-LDS staging).
// Pipeline:
//   1) ln_f16_kernel      : LN1(x) -> xn (f16, ws)
//   2) spatial_mix_kernel : shifted-window gather + per-head 49x49 mix via
//                           v_wmma_f32_16x16x32_f16 + residual -> d_out (x_mid)
//   3) ln_f16_kernel      : LN2(x_mid) -> h2n (f16, ws)
//   4) mlp_gemm_kernel<0> : h2n @ w1 + bb1, exact GELU -> act (f16, ws)
//   5) mlp_gemm_kernel<1> : act @ w2 + bb2 + x_mid -> d_out (f32)
// GEMMs: 128x128 block tile, double-buffered LDS, A-tiles staged with
// global_load_async_to_lds_b128 (ASYNCcnt), B-tiles f32->f16 transposed.
// Workspace: act region (411MB, aliased with xn) + h2n (103MB) => ~514MB.
// ---------------------------------------------------------------------------

typedef __attribute__((ext_vector_type(16))) _Float16 v16h;
typedef __attribute__((ext_vector_type(8)))  float    v8f;

#define WMMA_F32_16x16x32_F16(a, b, c) \
  __builtin_amdgcn_wmma_f32_16x16x32_f16(false, (a), false, (b), (short)0, (c), false, false)

// Static problem config
constexpr int B_  = 32, H_ = 56, W_ = 56, C_ = 512;
constexpr int NH_ = 16, HD_ = 32, WS_ = 7, WW_ = 49;
constexpr int PL_ = 4,  PT_ = 4, nH_ = 9, nW_ = 9;
constexpr int HID_ = 2048;
constexpr int MTOK = B_ * H_ * W_;    // 100352 tokens
constexpr int NWIN = B_ * nH_ * nW_;  // 2592 windows

union Frag16 { v16h v; uint4 q[2]; };

// Generic pointer to a __shared__ object: low 32 bits are the wave-relative
// LDS offset (flat LDS aperture puts the DS offset in addr[31:0]).
__device__ __forceinline__ uint32_t lds_off(const void* p) {
  return (uint32_t)(uintptr_t)p;
}

// 16B global -> LDS async copy (per-lane addresses, tracked by ASYNCcnt).
__device__ __forceinline__ void async_copy_b128(uint32_t lds_byte, const void* gaddr) {
  asm volatile("global_load_async_to_lds_b128 %0, %1, off"
               :: "v"(lds_byte), "v"(gaddr)
               : "memory");
}

__device__ __forceinline__ void wait_asynccnt0() {
  asm volatile("s_wait_asynccnt 0x0" ::: "memory");
}

// ---------------------------------------------------------------------------
// LayerNorm over C=512, one block (256 threads) per token, f16 output.
// ---------------------------------------------------------------------------
__global__ __launch_bounds__(256) void ln_f16_kernel(
    const float* __restrict__ x, const float* __restrict__ g,
    const float* __restrict__ b, _Float16* __restrict__ out) {
  const int tid  = threadIdx.x;
  const int lane = tid & 31, wid = tid >> 5;
  const size_t base = (size_t)blockIdx.x * C_;

  float v0 = x[base + tid];
  float v1 = x[base + tid + 256];
  float s  = v0 + v1;
  float ss = v0 * v0 + v1 * v1;
  for (int o = 16; o > 0; o >>= 1) {   // wave32 reduction
    s  += __shfl_xor(s, o);
    ss += __shfl_xor(ss, o);
  }
  __shared__ float rs[8], rq[8];
  if (lane == 0) { rs[wid] = s; rq[wid] = ss; }
  __syncthreads();
  float ts = 0.f, tq = 0.f;
  for (int i = 0; i < 8; ++i) { ts += rs[i]; tq += rq[i]; }
  const float mean = ts * (1.0f / C_);
  const float var  = tq * (1.0f / C_) - mean * mean;
  const float rstd = rsqrtf(var + 1e-5f);

  out[base + tid]       = (_Float16)((v0 - mean) * rstd * g[tid]       + b[tid]);
  out[base + tid + 256] = (_Float16)((v1 - mean) * rstd * g[tid + 256] + b[tid + 256]);
}

// ---------------------------------------------------------------------------
// Spatial mix: one block per window (256 threads = 8 waves).
// Per head h: out[i,d] = sum_j sw[h,i,j] * xn_gathered[j, h*32+d]  (i,j<49)
// M=49->64 (4 tiles), K=49->64 (2 steps), N=32 (2 tiles): wave w covers
// (mi = w>>1, ni = w&1). Result scattered back with crop + residual add.
// ---------------------------------------------------------------------------
__global__ __launch_bounds__(256) void spatial_mix_kernel(
    const float* __restrict__ x, const _Float16* __restrict__ xn,
    const float* __restrict__ sw, const float* __restrict__ sb,
    float* __restrict__ out) {
  __shared__ __align__(16) _Float16 Af[64 * 64];   // sw[head], zero-padded
  __shared__ __align__(16) _Float16 Bt[32 * 64];   // [d][j] transposed window tile
  __shared__ int jtok[64];                         // window pos -> token (or -1)

  const int tid = threadIdx.x, lane = tid & 31, wid = tid >> 5;
  const int win  = blockIdx.x;
  const int bb   = win / (nH_ * nW_);
  const int rem  = win % (nH_ * nW_);
  const int wh   = rem / nW_, wwin = rem % nW_;

  if (tid < 64) {
    int j = tid, v = -1;
    if (j < WW_) {
      int ph = wh * WS_ + j / WS_ - PT_;
      int pw = wwin * WS_ + j % WS_ - PL_;
      if (ph >= 0 && ph < H_ && pw >= 0 && pw < W_)
        v = (bb * H_ + ph) * W_ + pw;
    }
    jtok[j] = v;
  }
  __syncthreads();

  const int mi   = wid >> 1;            // M tile 0..3
  const int ni   = wid & 1;             // N tile 0..1
  const int lrow = lane & 15;
  const int aoff = (lane & 16) ? 8  : 0;  // A-frag K sub-offset per lane half
  const int boff = (lane & 16) ? 8  : 0;  // B-frag K sub-offset per lane half

  for (int head = 0; head < NH_; ++head) {
    // Stage A = sw[head] as 64x64 f16 (zero pad beyond 49)
    {
      const int i  = tid >> 2;
      const int j0 = (tid & 3) * 16;
      const float* swh = sw + ((size_t)head * WW_ + i) * WW_;
      for (int q = 0; q < 16; ++q) {
        int j = j0 + q;
        float v = (i < WW_ && j < WW_) ? swh[j] : 0.f;
        Af[i * 64 + j] = (_Float16)v;
      }
    }
    // Stage B transposed: Bt[d][j] = xn[token(j)][head*32+d] (zero for pad)
    if (tid < 64) {
      const int j = tid, t = jtok[j];
      if (t >= 0) {
        const _Float16* src = xn + (size_t)t * C_ + head * HD_;
        for (int d = 0; d < HD_; ++d) Bt[d * 64 + j] = src[d];
      } else {
        for (int d = 0; d < HD_; ++d) Bt[d * 64 + j] = (_Float16)0.f;
      }
    }
    __syncthreads();

    v8f acc = {};
    for (int ks = 0; ks < 64; ks += 32) {
      Frag16 fa, fb;
      const int arow = mi * 16 + lrow;
      fa.q[0] = *(const uint4*)&Af[arow * 64 + ks + aoff];
      fa.q[1] = *(const uint4*)&Af[arow * 64 + ks + 16 + aoff];
      const int bcol = ni * 16 + lrow;
      fb.q[0] = *(const uint4*)&Bt[bcol * 64 + ks + boff];
      fb.q[1] = *(const uint4*)&Bt[bcol * 64 + ks + boff + 16];
      acc = WMMA_F32_16x16x32_F16(fa.v, fb.v, acc);
    }

    // Epilogue: bias + residual, scatter through shifted-window crop
    const int d = ni * 16 + lrow;
    for (int r = 0; r < 8; ++r) {
      const int i = mi * 16 + r + ((lane & 16) ? 8 : 0);
      if (i < WW_) {
        const int t = jtok[i];
        if (t >= 0) {
          const size_t idx = (size_t)t * C_ + head * HD_ + d;
          out[idx] = x[idx] + acc[r] + sb[head * WW_ + i];
        }
      }
    }
    __syncthreads();  // LDS reused next head
  }
}

// ---------------------------------------------------------------------------
// Tiled GEMM: C[M,N] = A[M,K](f16) @ W[K,N](f32, cast to f16 in LDS).
// Block tile 128x128, 8 waves each owning 64x32 (4x2 WMMA accumulators).
// Double-buffered LDS; A staged via global_load_async_to_lds_b128 (ASYNCcnt);
// B staged f32->f16 with transpose. One barrier per K-step:
//   wait_asynccnt(0); barrier; issue stage i+1 into other buffer; WMMA on i.
// MODE 0: + bias, exact GELU, write f16.  MODE 1: + bias + residual, f32.
// M=100352, N in {2048,512}, K in {512,2048} -> all tiles exact, no guards.
// ---------------------------------------------------------------------------
template <int MODE, int N_, int K_>
__global__ __launch_bounds__(256) void mlp_gemm_kernel(
    const _Float16* __restrict__ A, const float* __restrict__ Wt,
    const float* __restrict__ bias, _Float16* __restrict__ out16,
    float* __restrict__ outf) {
  __shared__ __align__(16) _Float16 As[2][128 * 32];  // [m][k]
  __shared__ __align__(16) _Float16 Bs[2][128 * 32];  // [n][k] (transposed)

  const int tid = threadIdx.x, lane = tid & 31, wid = tid >> 5;
  constexpr int NB = N_ / 128;
  constexpr int NK = K_ / 32;
  const int nbase = (blockIdx.x % NB) * 128;
  const int mbase = (blockIdx.x / NB) * 128;
  const int m_off = (wid >> 2) * 64;
  const int n_off = (wid & 3) * 32;
  const int lrow  = lane & 15;
  const int aoff  = (lane & 16) ? 8  : 0;
  const int boff  = (lane & 16) ? 8  : 0;

  const uint32_t as_base = lds_off(&As[0][0]);

  // Per-thread A-staging geometry: 512 x 16B segments, 2 per thread.
  const int sg0_m = tid >> 2, sg0_k = (tid & 3) * 8;          // seg = tid
  const int sg1_m = (tid + 256) >> 2, sg1_k = ((tid + 256) & 3) * 8;

  v8f acc[4][2] = {};

  auto issueA = [&](int bufi, int kb) {
    const uint32_t lb = as_base + (uint32_t)bufi * (128 * 32 * 2);
    async_copy_b128(lb + (uint32_t)(sg0_m * 32 + sg0_k) * 2,
                    &A[(size_t)(mbase + sg0_m) * K_ + kb + sg0_k]);
    async_copy_b128(lb + (uint32_t)(sg1_m * 32 + sg1_k) * 2,
                    &A[(size_t)(mbase + sg1_m) * K_ + kb + sg1_k]);
  };
  auto stageB = [&](int bufi, int kb) {
    for (int r = 0; r < 16; ++r) {
      const int idx = tid + r * 256;
      const int kk = idx >> 7, nn = idx & 127;
      Bs[bufi][nn * 32 + kk] = (_Float16)Wt[(size_t)(kb + kk) * N_ + nbase + nn];
    }
  };

  issueA(0, 0);
  stageB(0, 0);

  for (int it = 0; it < NK; ++it) {
    const int bufi = it & 1;
    wait_asynccnt0();     // this wave's async A copies for stage `it` landed
    __syncthreads();      // + everyone's B ds_stores (compiler waits DScnt)

    if (it + 1 < NK) {    // refill the other buffer while we compute
      issueA(bufi ^ 1, (it + 1) * 32);
      stageB(bufi ^ 1, (it + 1) * 32);
      __builtin_prefetch(&Wt[(size_t)((it + 1) * 32 + (tid & 31)) * N_ + nbase + (tid >> 5) * 16], 0, 0);
    }

    Frag16 fa[4], fb[2];
    const _Float16* as = As[bufi];
    const _Float16* bs = Bs[bufi];
    for (int im = 0; im < 4; ++im) {
      const int row = m_off + im * 16 + lrow;
      fa[im].q[0] = *(const uint4*)&as[row * 32 + aoff];
      fa[im].q[1] = *(const uint4*)&as[row * 32 + 16 + aoff];
    }
    for (int in_ = 0; in_ < 2; ++in_) {
      const int col = n_off + in_ * 16 + lrow;
      fb[in_].q[0] = *(const uint4*)&bs[col * 32 + boff];
      fb[in_].q[1] = *(const uint4*)&bs[col * 32 + boff + 16];
    }
    for (int im = 0; im < 4; ++im)
      for (int in_ = 0; in_ < 2; ++in_)
        acc[im][in_] = WMMA_F32_16x16x32_F16(fa[im].v, fb[in_].v, acc[im][in_]);
    // no trailing barrier: next iteration's top barrier retires these reads
  }

  // Epilogue (C/D layout: VGPR r -> M = r + 8*(lane>=16); N = lane%16)
  for (int im = 0; im < 4; ++im)
    for (int in_ = 0; in_ < 2; ++in_) {
      const int n = nbase + n_off + in_ * 16 + lrow;
      const float bn = bias[n];
      for (int r = 0; r < 8; ++r) {
        const int m = mbase + m_off + im * 16 + r + ((lane & 16) ? 8 : 0);
        float v = acc[im][in_][r] + bn;
        const size_t o = (size_t)m * N_ + n;
        if (MODE == 0) {
          v = 0.5f * v * (1.0f + erff(v * 0.70710678118654752f));  // exact GELU
          out16[o] = (_Float16)v;
        } else {
          outf[o] = outf[o] + v;  // residual: d_out already holds x_mid
        }
      }
    }
}

// ---------------------------------------------------------------------------
// Launch
// ---------------------------------------------------------------------------
extern "C" void kernel_launch(void* const* d_in, const int* in_sizes, int n_in,
                              void* d_out, int out_size, void* d_ws, size_t ws_size,
                              hipStream_t stream) {
  (void)in_sizes; (void)n_in; (void)out_size; (void)ws_size;
  const float* x   = (const float*)d_in[0];
  const float* g1  = (const float*)d_in[1];
  const float* b1  = (const float*)d_in[2];
  const float* sw  = (const float*)d_in[3];
  const float* sb  = (const float*)d_in[4];
  const float* g2  = (const float*)d_in[5];
  const float* b2  = (const float*)d_in[6];
  const float* w1  = (const float*)d_in[7];
  const float* bb1 = (const float*)d_in[8];
  const float* w2  = (const float*)d_in[9];
  const float* bb2 = (const float*)d_in[10];
  float* out = (float*)d_out;

  char* ws = (char*)d_ws;
  const size_t ACT_BYTES = (size_t)MTOK * HID_ * sizeof(_Float16);  // 411MB
  _Float16* xn  = (_Float16*)ws;               // LN1 output (dead after stage 2)
  _Float16* act = (_Float16*)ws;               // aliases xn
  _Float16* h2n = (_Float16*)(ws + ACT_BYTES); // LN2 output (103MB)

  ln_f16_kernel<<<MTOK, 256, 0, stream>>>(x, g1, b1, xn);
  spatial_mix_kernel<<<NWIN, 256, 0, stream>>>(x, xn, sw, sb, out);
  ln_f16_kernel<<<MTOK, 256, 0, stream>>>(out, g2, b2, h2n);
  mlp_gemm_kernel<0, HID_, C_>
      <<<(MTOK / 128) * (HID_ / 128), 256, 0, stream>>>(h2n, w1, bb1, act, nullptr);
  mlp_gemm_kernel<1, C_, HID_>
      <<<(MTOK / 128) * (C_ / 128), 256, 0, stream>>>(act, w2, bb2, nullptr, out);
}
